// SelfAttention_45071386804277
// MI455X (gfx1250) — matrix operand order repaired and checked by reading, not compiled, not run on hardware
//
#include <hip/hip_runtime.h>
#include <hip/hip_bf16.h>
#include <stdint.h>

// ---------------------------------------------------------------------------
// Problem constants (from reference)
// ---------------------------------------------------------------------------
#define Bz    2
#define Sq    2048
#define Hd    2048
#define HQ    32
#define HKV   8
#define Gg    4
#define DKd   64
#define DVd   64
#define QKVO  3072            // HQ*DK + HKV*(DK+DV)
#define TQ    2048            // HQ*DK
#define TK    512             // HKV*DK
#define SCALE 0.125f          // 1/sqrt(64)

typedef __bf16 bf16_t;
typedef __attribute__((ext_vector_type(16))) __bf16 v16bf;
typedef __attribute__((ext_vector_type(8)))  __bf16 v8bf;
typedef __attribute__((ext_vector_type(4)))  __bf16 v4bf;
typedef __attribute__((ext_vector_type(8)))  float  v8f;

union BF16Frag {
    v16bf v;
    v8bf  h[2];
};
union PackB64 {
    bf16_t b[4];
    unsigned long long u;
};

__device__ __forceinline__ v8f wmma_bf16(const BF16Frag& a, const BF16Frag& b, v8f c) {
    // D(16x16,f32) = A(16x32,bf16) x B(32x16,bf16) + C
    return __builtin_amdgcn_wmma_f32_16x16x32_bf16(false, a.v, false, b.v,
                                                   (short)0, c, false, false);
}

// ---------------------------------------------------------------------------
// CDNA5 async global->LDS copy (16B per lane), tracked by ASYNCcnt.
// LDS byte offset = low 32 bits of the generic pointer (addrspacecast AS3->flat
// keeps the LDS offset in addr[31:0]; hardware adds the wave's LDS_BASE).
// ---------------------------------------------------------------------------
__device__ __forceinline__ void async_copy16(const bf16_t* gsrc, bf16_t* ldst) {
    unsigned lofs = (unsigned)(uintptr_t)ldst;
    asm volatile("global_load_async_to_lds_b128 %0, %1, off"
                 :: "v"(lofs), "v"(gsrc) : "memory");
}
__device__ __forceinline__ void wait_async_le4() {
    asm volatile("s_wait_asynccnt 0x4" ::: "memory");
}
__device__ __forceinline__ void wait_async_0() {
    asm volatile("s_wait_asynccnt 0x0" ::: "memory");
}

// ---------------------------------------------------------------------------
// Kernel 0a: f32 -> bf16 staging (x: row-major, K-contig, copied as-is)
// ---------------------------------------------------------------------------
__global__ __launch_bounds__(256) void cvt_f32_to_bf16(const float* __restrict__ in,
                                                       bf16_t* __restrict__ out, int n) {
    int stride = gridDim.x * blockDim.x * 4;
    for (int i = (blockIdx.x * blockDim.x + threadIdx.x) * 4; i < n; i += stride) {
        float4 f = *(const float4*)(in + i);
        v4bf o;
        o[0] = (bf16_t)f.x; o[1] = (bf16_t)f.y; o[2] = (bf16_t)f.z; o[3] = (bf16_t)f.w;
        *(v4bf*)(out + i) = o;
    }
}

// ---------------------------------------------------------------------------
// Kernel 0b: f32 W[K][N] -> bf16 W^T[N][K]  (LDS-tiled 32x32 transpose).
// ---------------------------------------------------------------------------
__global__ __launch_bounds__(256) void transpose_cvt_bf16(const float* __restrict__ in,
                                                          bf16_t* __restrict__ out,
                                                          int K, int N) {
    __shared__ bf16_t t[32][33];
    const int bx = blockIdx.x * 32;          // N
    const int by = blockIdx.y * 32;          // K
    const int tx = threadIdx.x & 31;
    const int ty = threadIdx.x >> 5;         // 0..7
#pragma unroll
    for (int i = 0; i < 32; i += 8)
        t[ty + i][tx] = (bf16_t)in[(size_t)(by + ty + i) * N + bx + tx];
    __syncthreads();
#pragma unroll
    for (int i = 0; i < 32; i += 8)
        out[(size_t)(bx + ty + i) * K + by + tx] = t[tx][ty + i];
}

// ---------------------------------------------------------------------------
// Tiled bf16 WMMA GEMM:  C[M,N] = A[M,K] * Bt[N,K]^T  (+ bias)
// Block tile 128x128, BK=32.  256 threads = 8 waves (2M x 4N); wave = 64x32.
// Double-buffered LDS, tiles filled by async global->LDS B128 copies.
// ---------------------------------------------------------------------------
#define GBM 128
#define GBN 128
#define GBK 32
#define GSTR 40   // 32 + 8 pad -> 80B rows: conflict-free, 16B aligned

template <typename OutT, bool HAS_BIAS>
__global__ __launch_bounds__(256)
void gemm_wmma(const bf16_t* __restrict__ A, const bf16_t* __restrict__ Bt,
               OutT* __restrict__ C, const float* __restrict__ bias,
               int M, int N, int K) {
    __shared__ __attribute__((aligned(16))) bf16_t As [2][GBM][GSTR]; // M-major, K contig
    __shared__ __attribute__((aligned(16))) bf16_t BsT[2][GBN][GSTR]; // N-major, K contig

    const int tid  = threadIdx.x;
    const int lane = tid & 31;
    const int wave = tid >> 5;
    const int hi   = lane >> 4;     // half-wave select (A/B layout K split)
    const int l15  = lane & 15;

    const int bm0 = blockIdx.y * GBM;
    const int bn0 = blockIdx.x * GBN;
    const int wm  = (wave >> 2) * 64;   // 0 / 64
    const int wn  = (wave & 3) * 32;    // 0 / 32 / 64 / 96

    // per-thread copy slot: 128 rows x 32 K split as (row = tid/2, 16-col seg)
    const int cr = tid >> 1;
    const int cc = (tid & 1) * 16;
    const bf16_t* aSrc = A  + (size_t)(bm0 + cr) * K + cc;
    const bf16_t* bSrc = Bt + (size_t)(bn0 + cr) * K + cc;

    v8f acc[4][2];
#pragma unroll
    for (int i = 0; i < 4; ++i)
#pragma unroll
        for (int j = 0; j < 2; ++j) acc[i][j] = (v8f){};

    const int nk = K / GBK;

    // prefetch tile 0 (4 async wave-instructions: 2 for A, 2 for B)
    async_copy16(aSrc,     &As [0][cr][cc]);
    async_copy16(aSrc + 8, &As [0][cr][cc + 8]);
    async_copy16(bSrc,     &BsT[0][cr][cc]);
    async_copy16(bSrc + 8, &BsT[0][cr][cc + 8]);

    for (int kt = 0; kt < nk; ++kt) {
        const int buf = kt & 1;
        if (kt + 1 < nk) {
            const bf16_t* an = aSrc + (size_t)(kt + 1) * GBK;
            const bf16_t* bn = bSrc + (size_t)(kt + 1) * GBK;
            async_copy16(an,     &As [buf ^ 1][cr][cc]);
            async_copy16(an + 8, &As [buf ^ 1][cr][cc + 8]);
            async_copy16(bn,     &BsT[buf ^ 1][cr][cc]);
            async_copy16(bn + 8, &BsT[buf ^ 1][cr][cc + 8]);
            wait_async_le4();           // older 4 (tile kt) complete, in order
        } else {
            wait_async_0();
        }
        __syncthreads();                // tile kt visible to all waves

        BF16Frag afr[4], bfr[2];
#pragma unroll
        for (int i = 0; i < 4; ++i) {
            int row = wm + i * 16 + l15;
            int kb  = hi * 8;           // lanes 0-15: K 0..7/16..23; 16-31: 8..15/24..31
            afr[i].h[0] = *(const v8bf*)&As[buf][row][kb];
            afr[i].h[1] = *(const v8bf*)&As[buf][row][kb + 16];
        }
#pragma unroll
        for (int j = 0; j < 2; ++j) {
            int col = wn + j * 16 + l15;
            int kb  = hi * 8;           // lanes 0-15: K 0..7/16..23; 16-31: 8..15/24..31
            bfr[j].h[0] = *(const v8bf*)&BsT[buf][col][kb];
            bfr[j].h[1] = *(const v8bf*)&BsT[buf][col][kb + 16];
        }
#pragma unroll
        for (int i = 0; i < 4; ++i)
#pragma unroll
            for (int j = 0; j < 2; ++j)
                acc[i][j] = wmma_bf16(afr[i], bfr[j], acc[i][j]);
        __syncthreads();                // all reads of buf done before its refill
    }

    // epilogue: D layout M = r + 8*hi, N = l15
#pragma unroll
    for (int i = 0; i < 4; ++i)
#pragma unroll
        for (int j = 0; j < 2; ++j) {
            int col = bn0 + wn + j * 16 + l15;
            float bv = HAS_BIAS ? bias[col] : 0.0f;
#pragma unroll
            for (int r = 0; r < 8; ++r) {
                int row = bm0 + wm + i * 16 + r + hi * 8;
                C[(size_t)row * N + col] = (OutT)(acc[i][j][r] + bv);
            }
        }
}

// ---------------------------------------------------------------------------
// Kernel 2: causal flash attention, GQA-fused: one block = all G=4 query heads
// of one kv-head x 64 queries, sharing each K/V tile (4x less K/V traffic).
// grid = (S/64, B*HKV).  256 threads = 8 waves; wave w -> head (w>>1),
// query rows (w&1)*32..+31 processed as two sequential 16-row chunks.
// Q/K tiles arrive via async global->LDS; V transposed with packed b64 stores.
// ---------------------------------------------------------------------------
#define FSTR 72   // 64 + 8 pad -> 144B rows: conflict-free, 16B aligned

__global__ __launch_bounds__(256)
void flash_attn(const bf16_t* __restrict__ qkv, bf16_t* __restrict__ attn) {
    __shared__ __attribute__((aligned(16))) bf16_t Qs [256][FSTR]; // 4 heads x 64 q-rows
    __shared__ __attribute__((aligned(16))) bf16_t Ks [64][FSTR];  // key-major, d contig
    __shared__ __attribute__((aligned(16))) bf16_t VsT[64][FSTR];  // dv-major, key contig
    __shared__ __attribute__((aligned(16))) bf16_t Ps [256][FSTR]; // wave-private 32-row slabs

    const int tid  = threadIdx.x;
    const int lane = tid & 31;
    const int wave = tid >> 5;          // 0..7
    const int hi   = lane >> 4;
    const int l15  = lane & 15;

    const int qb = blockIdx.x;          // query block (64 rows)
    const int bn = blockIdx.y;          // b*HKV + n
    const int b  = bn >> 3;
    const int n  = bn & 7;              // kv head

    const size_t base = (size_t)b * Sq * QKVO;
    const int q0   = qb * 64;
    const int koff = TQ + n * DKd;
    const int voff = TQ + TK + n * DVd;

    const int g    = wave >> 1;         // query head within group for this wave
    const int hq   = n * Gg + g;
    const int wr   = (wave & 1) * 32;   // row base within the head's 64 rows

    // ---- async Q copy: 4 heads x 64 rows; thread t -> Qs row t ----
    {
        int gq = tid >> 6, ql = tid & 63;
        const bf16_t* src = qkv + base + (size_t)(q0 + ql) * QKVO + (n * Gg + gq) * DKd;
#pragma unroll
        for (int i = 0; i < 8; ++i)
            async_copy16(src + i * 8, &Qs[tid][i * 8]);
    }

    v8f o[2][4];
#pragma unroll
    for (int c = 0; c < 2; ++c)
#pragma unroll
        for (int j = 0; j < 4; ++j) o[c][j] = (v8f){};
    float mrow[2][8], lrow[2][8];
#pragma unroll
    for (int c = 0; c < 2; ++c)
#pragma unroll
        for (int r = 0; r < 8; ++r) { mrow[c][r] = -3.0e38f; lrow[c][r] = 0.0f; }

    for (int jb = 0; jb <= qb; ++jb) {
        __syncthreads();   // previous K/V tile fully consumed before refill
        // async K tile: thread t -> row t/4, 16-col segment (t&3)*16
        {
            int r = tid >> 2, c0 = (tid & 3) * 16;
            const bf16_t* ksrc = qkv + base + (size_t)(jb * 64 + r) * QKVO + koff + c0;
            async_copy16(ksrc,     &Ks[r][c0]);
            async_copy16(ksrc + 8, &Ks[r][c0 + 8]);
        }
        // V tile: 4-key x 4-dv micro-tile per thread, packed b64 transposed stores
        {
            int k0  = (tid & 15) * 4;        // key quad
            int dv0 = (tid >> 4) * 4;        // dv quad
            v4bf vq[4];
#pragma unroll
            for (int ki = 0; ki < 4; ++ki)
                vq[ki] = *(const v4bf*)(qkv + base +
                         (size_t)(jb * 64 + k0 + ki) * QKVO + voff + dv0);
#pragma unroll
            for (int e = 0; e < 4; ++e) {
                PackB64 p;
                p.b[0] = vq[0][e]; p.b[1] = vq[1][e];
                p.b[2] = vq[2][e]; p.b[3] = vq[3][e];
                *(unsigned long long*)&VsT[dv0 + e][k0] = p.u;
            }
        }
        wait_async_0();
        __syncthreads();

        const bool diag = (jb == qb);
#pragma unroll
        for (int c = 0; c < 2; ++c) {
            const int qrow0 = g * 64 + wr + c * 16;    // Qs row base, this chunk
            const int prow0 = wave * 32 + c * 16;      // wave-private Ps slab

            // ---- S = Q @ K^T  (16 q-rows x 64 keys) ----
            v8f s[4];
#pragma unroll
            for (int j = 0; j < 4; ++j) s[j] = (v8f){};
#pragma unroll
            for (int k0 = 0; k0 < DKd; k0 += 32) {
                BF16Frag a;
                int row = qrow0 + l15;
                int kb  = k0 + hi * 8;
                a.h[0] = *(const v8bf*)&Qs[row][kb];
                a.h[1] = *(const v8bf*)&Qs[row][kb + 16];
#pragma unroll
                for (int j = 0; j < 4; ++j) {
                    BF16Frag bb;
                    int key = j * 16 + l15;
                    int kh  = k0 + hi * 8;
                    bb.h[0] = *(const v8bf*)&Ks[key][kh];
                    bb.h[1] = *(const v8bf*)&Ks[key][kh + 16];
                    s[j] = wmma_bf16(a, bb, s[j]);
                }
            }

            // ---- scale + causal mask + online softmax (row = r + 8*hi) ----
            float corr[8];
#pragma unroll
            for (int r = 0; r < 8; ++r) {
                int   qrow = q0 + wr + c * 16 + r + hi * 8;
                float mx = -3.0e38f;
#pragma unroll
                for (int j = 0; j < 4; ++j) {
                    float v = s[j][r] * SCALE;
                    if (diag && (jb * 64 + j * 16 + l15) > qrow) v = -3.0e38f;
                    s[j][r] = v;
                    mx = fmaxf(mx, v);
                }
#pragma unroll
                for (int off = 1; off < 16; off <<= 1)   // reduce within 16-lane half
                    mx = fmaxf(mx, __shfl_xor(mx, off, 32));
                float mnew = fmaxf(mrow[c][r], mx);
                float cf   = __expf(mrow[c][r] - mnew);
                mrow[c][r] = mnew;
                corr[r]    = cf;
                float sum = 0.0f;
#pragma unroll
                for (int j = 0; j < 4; ++j) {
                    float p = __expf(s[j][r] - mnew);
                    s[j][r] = p;
                    sum += p;
                }
#pragma unroll
                for (int off = 1; off < 16; off <<= 1)
                    sum += __shfl_xor(sum, off, 32);
                lrow[c][r] = lrow[c][r] * cf + sum;
            }

            // stage P (D-layout -> A-layout via wave-private LDS), rescale O
#pragma unroll
            for (int j = 0; j < 4; ++j)
#pragma unroll
                for (int r = 0; r < 8; ++r)
                    Ps[prow0 + r + hi * 8][j * 16 + l15] = (bf16_t)s[j][r];
#pragma unroll
            for (int j = 0; j < 4; ++j)
#pragma unroll
                for (int r = 0; r < 8; ++r) o[c][j][r] *= corr[r];

            // ---- O += P @ V ----
#pragma unroll
            for (int k0 = 0; k0 < 64; k0 += 32) {
                BF16Frag a;
                int row = prow0 + l15;
                int kb  = k0 + hi * 8;
                a.h[0] = *(const v8bf*)&Ps[row][kb];
                a.h[1] = *(const v8bf*)&Ps[row][kb + 16];
#pragma unroll
                for (int j = 0; j < 4; ++j) {
                    BF16Frag bb;
                    int dv = j * 16 + l15;
                    int kh = k0 + hi * 8;
                    bb.h[0] = *(const v8bf*)&VsT[dv][kh];
                    bb.h[1] = *(const v8bf*)&VsT[dv][kh + 16];
                    o[c][j] = wmma_bf16(a, bb, o[c][j]);
                }
            }
        }
    }

    // ---- epilogue: normalize, store bf16 attn[(b,s), hq*64 + dv] ----
#pragma unroll
    for (int c = 0; c < 2; ++c)
#pragma unroll
        for (int r = 0; r < 8; ++r) {
            float inv  = 1.0f / lrow[c][r];
            int   qrow = q0 + wr + c * 16 + r + hi * 8;
            size_t ob  = ((size_t)b * Sq + qrow) * (size_t)(HQ * DVd) + hq * DVd;
#pragma unroll
            for (int j = 0; j < 4; ++j)
                attn[ob + j * 16 + l15] = (bf16_t)(o[c][j][r] * inv);
        }
}

// ---------------------------------------------------------------------------
// Launcher
// ---------------------------------------------------------------------------
extern "C" void kernel_launch(void* const* d_in, const int* in_sizes, int n_in,
                              void* d_out, int out_size, void* d_ws, size_t ws_size,
                              hipStream_t stream) {
    const float* x     = (const float*)d_in[0];  // (B,S,H)
    const float* Wqkv  = (const float*)d_in[1];  // (H, 3072)
    const float* Wout  = (const float*)d_in[2];  // (2048, H)
    const float* b_out = (const float*)d_in[3];  // (H,)
    // d_in[4] = mask: causal tril, reproduced analytically in-kernel.
    float* out = (float*)d_out;

    const int nx    = Bz * Sq * Hd;        // 8388608
    const int nwqkv = Hd * QKVO;           // 6291456
    const int nwout = (HQ * DVd) * Hd;     // 4194304
    const int nqkv  = Bz * Sq * QKVO;      // 12582912

    bf16_t* xbf     = (bf16_t*)d_ws;
    bf16_t* wqkvT   = xbf + nx;            // bf16 Wqkv^T [3072][2048]
    bf16_t* woutT   = wqkvT + nwqkv;       // bf16 Wout^T [2048][2048]
    bf16_t* qkvbf   = woutT + nwout;
    bf16_t* attnbf  = qkvbf + nqkv;

    // 0) stage operands to bf16 (weights transposed to N-major/K-contig)
    cvt_f32_to_bf16<<<4096, 256, 0, stream>>>(x, xbf, nx);
    {
        dim3 g1(QKVO / 32, Hd / 32);
        transpose_cvt_bf16<<<g1, 256, 0, stream>>>(Wqkv, wqkvT, Hd, QKVO);
        dim3 g2(Hd / 32, (HQ * DVd) / 32);
        transpose_cvt_bf16<<<g2, 256, 0, stream>>>(Wout, woutT, HQ * DVd, Hd);
    }

    // 1) qkv = x @ Wqkv   : M=4096, N=3072, K=2048 -> bf16
    {
        dim3 grid(QKVO / GBN, (Bz * Sq) / GBM);
        gemm_wmma<bf16_t, false><<<grid, 256, 0, stream>>>(
            xbf, wqkvT, qkvbf, nullptr, Bz * Sq, QKVO, Hd);
    }

    // 2) causal flash attention (GQA-fused) -> attn bf16
    {
        dim3 grid(Sq / 64, Bz * HKV);
        flash_attn<<<grid, 256, 0, stream>>>(qkvbf, attnbf);
    }

    // 3) out = attn @ Wout + b_out : M=4096, N=2048, K=2048 -> f32
    {
        dim3 grid(Hd / GBN, (Bz * Sq) / GBM);
        gemm_wmma<float, true><<<grid, 256, 0, stream>>>(
            attnbf, woutT, out, b_out, Bz * Sq, Hd, HQ * DVd);
    }
    (void)in_sizes; (void)n_in; (void)out_size; (void)ws_size;
}
